// TimeMix_15865609191750
// MI455X (gfx1250) — matrix operand order repaired and checked by reading, not compiled, not run on hardware
//
#include <hip/hip_runtime.h>

typedef __bf16 bf16_t;
typedef bf16_t v16bf __attribute__((ext_vector_type(16)));
typedef float  v8f   __attribute__((ext_vector_type(8)));

#define BB_DIM 4
#define T_DIM  2048
#define C_DIM  2048
#define M_DIM  (BB_DIM * T_DIM)   // 8192

#if __has_builtin(__builtin_amdgcn_global_load_async_to_lds_b128)
#define USE_ASYNC 1
#else
#define USE_ASYNC 0
#endif

typedef int v4i_async __attribute__((vector_size(16)));
typedef __attribute__((address_space(1))) v4i_async* async_gptr;
typedef __attribute__((address_space(3))) v4i_async* async_lptr;

static __device__ __forceinline__ void cp16_async(bf16_t* dst, const bf16_t* src) {
#if USE_ASYNC
    __builtin_amdgcn_global_load_async_to_lds_b128(
        (async_gptr)src, (async_lptr)dst, 0, 0);
#else
    *(uint4*)dst = *(const uint4*)src;
#endif
}

static __device__ __forceinline__ void wait_async_all() {
#if USE_ASYNC
#if __has_builtin(__builtin_amdgcn_s_wait_asynccnt)
    __builtin_amdgcn_s_wait_asynccnt(0);
#else
    asm volatile("s_wait_asynccnt 0x0" ::: "memory");
#endif
#endif
}

static __device__ __forceinline__ bf16_t f2bf(float f) {
    unsigned u = __float_as_uint(f);
    u += 0x7FFFu + ((u >> 16) & 1u);          // round-to-nearest-even
    unsigned short s = (unsigned short)(u >> 16);
    bf16_t out;
    __builtin_memcpy(&out, &s, 2);
    return out;
}

// ---------------------------------------------------------------- weights f32 -> bf16
__global__ void k_cvt_bf16(const float* __restrict__ in, bf16_t* __restrict__ out, int n4) {
    int i = blockIdx.x * blockDim.x + threadIdx.x;
    if (i >= n4) return;
    const float4 f = ((const float4*)in)[i];
    union { bf16_t h[4]; uint2 u; } t;
    t.h[0] = f2bf(f.x); t.h[1] = f2bf(f.y); t.h[2] = f2bf(f.z); t.h[3] = f2bf(f.w);
    ((uint2*)out)[i] = t.u;
}

// ---------------------------------------------------------------- fused triple GEMM
// k = mixK(x) @ Wk^T ; v = mixV(x) @ Wv^T ; rl = mixR(x) @ Wr^T
// Block tile 128(M) x 64(N), K-chunk 64, double-buffered LDS, async weight tiles,
// x loads pipelined through VGPRs around the WMMA block.
__global__ __launch_bounds__(256) void k_gemm3(
    const float* __restrict__ x,
    const float* __restrict__ tmk, const float* __restrict__ tmv,
    const float* __restrict__ tmr, const float* __restrict__ xxp,
    const bf16_t* __restrict__ Wk, const bf16_t* __restrict__ Wv, const bf16_t* __restrict__ Wr,
    float* __restrict__ outk, float* __restrict__ outv, float* __restrict__ outr)
{
    constexpr int K = C_DIM, N = C_DIM;
    constexpr int LDT = 72;                       // row stride: 144 B (16B-aligned, conflict-free)
    __shared__ __attribute__((aligned(16))) bf16_t As[2][3][128][LDT];
    __shared__ __attribute__((aligned(16))) bf16_t Bs[2][3][64][LDT];

    const int tid  = threadIdx.x;
    const int lane = tid & 31;
    const int wave = tid >> 5;
    const int wm = wave >> 1;                     // 0..3 (M)
    const int wn = wave & 1;                      // 0..1 (N)
    const int bM = blockIdx.x * 128;
    const int bN = blockIdx.y * 64;

    v8f acc[3][2][2];
    const v8f vzero = {0.f,0.f,0.f,0.f,0.f,0.f,0.f,0.f};
    #pragma unroll
    for (int g = 0; g < 3; ++g)
        #pragma unroll
        for (int i = 0; i < 2; ++i)
            #pragma unroll
            for (int j = 0; j < 2; ++j) acc[g][i][j] = vzero;

    const int cx = (tid & 15) * 4;                // x tile: 4 cols x 8 rows per thread
    const int rx = (tid >> 4) * 8;
    const int wrow  = tid >> 2;                   // weight tile mapping
    const int wcol8 = (tid & 3) * 8;
    const int half = lane >> 4;                   // WMMA lane decomposition (ISA 7.12.2)
    const int mr   = lane & 15;

    float4 xreg[8], ckr, cvr, crr, cxr;

    auto loadx = [&](int k0) {
        ckr = *(const float4*)&tmk[k0 + cx];
        cvr = *(const float4*)&tmv[k0 + cx];
        crr = *(const float4*)&tmr[k0 + cx];
        cxr = *(const float4*)&xxp[k0 + cx];
        #pragma unroll
        for (int r = 0; r < 8; ++r)
            xreg[r] = *(const float4*)&x[(size_t)(bM + rx + r) * K + k0 + cx];
    };

    auto xstore = [&](int buf) {
        const float kc[4] = {ckr.x, ckr.y, ckr.z, ckr.w};
        const float vc[4] = {cvr.x, cvr.y, cvr.z, cvr.w};
        const float rc[4] = {crr.x, crr.y, crr.z, crr.w};
        const float xc[4] = {cxr.x, cxr.y, cxr.z, cxr.w};
        #pragma unroll
        for (int r = 0; r < 8; ++r) {
            const float xs[4] = {xreg[r].x, xreg[r].y, xreg[r].z, xreg[r].w};
            union { bf16_t h[4]; uint2 u; } tk, tv, tr;
            #pragma unroll
            for (int c = 0; c < 4; ++c) {
                tk.h[c] = f2bf(xs[c] * kc[c] + (1.f - kc[c]) * xc[c]);
                tv.h[c] = f2bf(xs[c] * vc[c] + (1.f - vc[c]) * xc[c]);
                tr.h[c] = f2bf(xs[c] * rc[c] + (1.f - rc[c]) * xc[c]);
            }
            *(uint2*)&As[buf][0][rx + r][cx] = tk.u;
            *(uint2*)&As[buf][1][rx + r][cx] = tv.u;
            *(uint2*)&As[buf][2][rx + r][cx] = tr.u;
        }
    };

    auto issueB = [&](int buf, int k0) {
        const bf16_t* srcs[3] = {Wk, Wv, Wr};
        #pragma unroll
        for (int g = 0; g < 3; ++g) {
            const bf16_t* src = srcs[g] + (size_t)(bN + wrow) * K + k0;
            cp16_async(&Bs[buf][g][wrow][wcol8],      src + wcol8);
            cp16_async(&Bs[buf][g][wrow][wcol8 + 32], src + wcol8 + 32);
        }
    };

    auto domma = [&](int buf) {
        #pragma unroll
        for (int ks = 0; ks < 64; ks += 32) {
            #pragma unroll
            for (int g = 0; g < 3; ++g) {
                v16bf afrag[2], bfrag[2];
                #pragma unroll
                for (int i = 0; i < 2; ++i) {
                    const bf16_t* pa = &As[buf][g][wm * 32 + i * 16 + mr][ks + 8 * half];
                    ((uint4*)&afrag[i])[0] = *(const uint4*)pa;
                    ((uint4*)&afrag[i])[1] = *(const uint4*)(pa + 16);
                }
                #pragma unroll
                for (int j = 0; j < 2; ++j) {
                    const bf16_t* pb = &Bs[buf][g][wn * 32 + j * 16 + mr][ks + 8 * half];
                    ((uint4*)&bfrag[j])[0] = *(const uint4*)pb;
                    ((uint4*)&bfrag[j])[1] = *(const uint4*)(pb + 16);
                }
                #pragma unroll
                for (int i = 0; i < 2; ++i)
                    #pragma unroll
                    for (int j = 0; j < 2; ++j)
                        acc[g][i][j] = __builtin_amdgcn_wmma_f32_16x16x32_bf16(
                            false, afrag[i], false, bfrag[j],
                            (short)0, acc[g][i][j], false, false);
            }
        }
    };

    // ---- pipelined main loop (one barrier per K-chunk)
    loadx(0);
    issueB(0, 0);
    xstore(0);
    int buf = 0;
    for (int c = 0; c < K / 64; ++c) {
        const int k0n = (c + 1) * 64;
        wait_async_all();                         // chunk c's async B tiles landed (this wave)
        __syncthreads();                          // all waves: As stores + async B visible
        if (k0n < K) {
            issueB(buf ^ 1, k0n);                 // overlap next B tiles with compute
            loadx(k0n);                           // x loads in flight during WMMA
        }
        domma(buf);
        if (k0n < K) xstore(buf ^ 1);             // transform after compute; visible at next barrier
        buf ^= 1;
    }

    // ---- writeback (C/D layout: lanes 0-15 = N, vgpr e = M row; lanes 16-31 -> M+8)
    float* outs[3] = {outk, outv, outr};
    const int colN = lane & 15;
    const int rowO = (lane >> 4) * 8;
    #pragma unroll
    for (int g = 0; g < 3; ++g)
        #pragma unroll
        for (int i = 0; i < 2; ++i)
            #pragma unroll
            for (int j = 0; j < 2; ++j) {
                const int gn = bN + wn * 32 + j * 16 + colN;
                const int gm = bM + wm * 32 + i * 16 + rowO;
                #pragma unroll
                for (int e = 0; e < 8; ++e)
                    outs[g][(size_t)(gm + e) * N + gn] = acc[g][i][j][e];
            }
}

// ---------------------------------------------------------------- wkv + sigmoid -> bf16
__global__ void k_wkv(const float* __restrict__ kk, const float* __restrict__ vv,
                      const float* __restrict__ rl,
                      const float* __restrict__ tf, const float* __restrict__ aa,
                      const float* __restrict__ bbp, const float* __restrict__ pp,
                      bf16_t* __restrict__ z)
{
    const size_t i4 = (size_t)blockIdx.x * blockDim.x + threadIdx.x;
    const size_t base = i4 * 4;
    if (base >= (size_t)M_DIM * C_DIM) return;
    const int c = (int)(base & (C_DIM - 1));
    const int m = (int)(base >> 11);              // C = 2^11
    const bool first = ((m & (T_DIM - 1)) == 0);  // t == 0 row gets the real state

    const float4 k4 = *(const float4*)&kk[base];
    const float4 v4 = *(const float4*)&vv[base];
    const float4 r4 = *(const float4*)&rl[base];
    const float4 t4 = *(const float4*)&tf[c];
    float4 a4 = {0.f,0.f,0.f,0.f}, b4 = {0.f,0.f,0.f,0.f}, p4 = {0.f,0.f,0.f,0.f};
    if (first) {
        a4 = *(const float4*)&aa[c];
        b4 = *(const float4*)&bbp[c];
        p4 = *(const float4*)&pp[c];
    }
    const float ka[4] = {k4.x,k4.y,k4.z,k4.w};
    const float va[4] = {v4.x,v4.y,v4.z,v4.w};
    const float ra[4] = {r4.x,r4.y,r4.z,r4.w};
    const float ta[4] = {t4.x,t4.y,t4.z,t4.w};
    const float aaa[4] = {a4.x,a4.y,a4.z,a4.w};
    const float bba[4] = {b4.x,b4.y,b4.z,b4.w};
    const float ppa[4] = {p4.x,p4.y,p4.z,p4.w};
    union { bf16_t h[4]; uint2 u; } zo;
    #pragma unroll
    for (int j = 0; j < 4; ++j) {
        const float ww  = ta[j] + ka[j];
        const float qq  = fmaxf(ppa[j], ww);
        const float e1  = __expf(ppa[j] - qq);
        const float e2  = __expf(ww - qq);
        const float wkv = (e1 * aaa[j] + e2 * va[j]) / (e1 * bba[j] + e2);
        const float rr  = 1.f / (1.f + __expf(-ra[j]));
        zo.h[j] = f2bf(rr * wkv);
    }
    *(uint2*)&z[base] = zo.u;
}

// ---------------------------------------------------------------- output GEMM: out = z @ Wo^T
// Block tile 128x128, K-chunk 64, double-buffered, fully async tile loads.
__global__ __launch_bounds__(256) void k_gemm_out(
    const bf16_t* __restrict__ A, const bf16_t* __restrict__ Bw, float* __restrict__ out)
{
    constexpr int K = C_DIM, N = C_DIM;
    constexpr int LDT = 72;
    __shared__ __attribute__((aligned(16))) bf16_t As[2][128][LDT];
    __shared__ __attribute__((aligned(16))) bf16_t Bs[2][128][LDT];

    const int tid  = threadIdx.x;
    const int lane = tid & 31;
    const int wave = tid >> 5;
    const int wm = wave >> 1;                     // 0..3
    const int wn = wave & 1;                      // 0..1
    const int bM = blockIdx.x * 128;
    const int bN = blockIdx.y * 128;

    v8f acc[2][4];
    const v8f vzero = {0.f,0.f,0.f,0.f,0.f,0.f,0.f,0.f};
    #pragma unroll
    for (int i = 0; i < 2; ++i)
        #pragma unroll
        for (int j = 0; j < 4; ++j) acc[i][j] = vzero;

    const int ar = tid >> 1;                      // 0..127
    const int ac = (tid & 1) * 8;
    const int half = lane >> 4;
    const int mr   = lane & 15;

    auto issueAB = [&](int buf, int k0) {
        #pragma unroll
        for (int q = 0; q < 4; ++q) {
            cp16_async(&As[buf][ar][ac + q * 16],
                       &A[(size_t)(bM + ar) * K + k0 + ac + q * 16]);
            cp16_async(&Bs[buf][ar][ac + q * 16],
                       &Bw[(size_t)(bN + ar) * K + k0 + ac + q * 16]);
        }
    };

    issueAB(0, 0);
    int buf = 0;
    for (int c = 0; c < K / 64; ++c) {
        const int k0n = (c + 1) * 64;
        wait_async_all();
        __syncthreads();
        if (k0n < K) issueAB(buf ^ 1, k0n);
        #pragma unroll
        for (int ks = 0; ks < 64; ks += 32) {
            v16bf afrag[2];
            #pragma unroll
            for (int i = 0; i < 2; ++i) {
                const bf16_t* pa = &As[buf][wm * 32 + i * 16 + mr][ks + 8 * half];
                ((uint4*)&afrag[i])[0] = *(const uint4*)pa;
                ((uint4*)&afrag[i])[1] = *(const uint4*)(pa + 16);
            }
            #pragma unroll
            for (int j = 0; j < 4; ++j) {
                v16bf bfrag;
                const bf16_t* pb = &Bs[buf][wn * 64 + j * 16 + mr][ks + 8 * half];
                ((uint4*)&bfrag)[0] = *(const uint4*)pb;
                ((uint4*)&bfrag)[1] = *(const uint4*)(pb + 16);
                #pragma unroll
                for (int i = 0; i < 2; ++i)
                    acc[i][j] = __builtin_amdgcn_wmma_f32_16x16x32_bf16(
                        false, afrag[i], false, bfrag, (short)0, acc[i][j], false, false);
            }
        }
        buf ^= 1;
    }

    const int colN = lane & 15;
    const int rowO = (lane >> 4) * 8;
    #pragma unroll
    for (int i = 0; i < 2; ++i)
        #pragma unroll
        for (int j = 0; j < 4; ++j) {
            const int gn = bN + wn * 64 + j * 16 + colN;
            const int gm = bM + wm * 32 + i * 16 + rowO;
            #pragma unroll
            for (int e = 0; e < 8; ++e)
                out[(size_t)(gm + e) * N + gn] = acc[i][j][e];
        }
}

// ----------------------------------------------------------------
extern "C" void kernel_launch(void* const* d_in, const int* in_sizes, int n_in,
                              void* d_out, int out_size, void* d_ws, size_t ws_size,
                              hipStream_t stream) {
    const float* x   = (const float*)d_in[0];
    const float* tf  = (const float*)d_in[1];
    const float* tmk = (const float*)d_in[2];
    const float* tmv = (const float*)d_in[3];
    const float* tmr = (const float*)d_in[4];
    const float* xx  = (const float*)d_in[5];
    const float* aa  = (const float*)d_in[6];
    const float* bb  = (const float*)d_in[7];
    const float* pp  = (const float*)d_in[8];
    const float* wk  = (const float*)d_in[9];
    const float* wv  = (const float*)d_in[10];
    const float* wr  = (const float*)d_in[11];
    const float* wo  = (const float*)d_in[12];

    char* ws = (char*)d_ws;
    const size_t SZW = (size_t)C_DIM * C_DIM;     // 4Mi elements
    const size_t SZM = (size_t)M_DIM * C_DIM;     // 16Mi elements
    bf16_t* Wk = (bf16_t*)(ws);
    bf16_t* Wv = (bf16_t*)(ws + SZW * 2);
    bf16_t* Wr = (bf16_t*)(ws + SZW * 4);
    bf16_t* Wo = (bf16_t*)(ws + SZW * 6);
    float*  kb = (float*)(ws + SZW * 8);
    float*  vb = (float*)(ws + SZW * 8 + SZM * 4);
    float*  rb = (float*)(ws + SZW * 8 + SZM * 8);
    bf16_t* zb = (bf16_t*)(ws + SZW * 8 + SZM * 12);

    const int n4w = (int)(SZW / 4);
    const int cvtBlocks = (n4w + 255) / 256;
    k_cvt_bf16<<<cvtBlocks, 256, 0, stream>>>(wk, Wk, n4w);
    k_cvt_bf16<<<cvtBlocks, 256, 0, stream>>>(wv, Wv, n4w);
    k_cvt_bf16<<<cvtBlocks, 256, 0, stream>>>(wr, Wr, n4w);
    k_cvt_bf16<<<cvtBlocks, 256, 0, stream>>>(wo, Wo, n4w);

    dim3 g3(M_DIM / 128, C_DIM / 64);
    k_gemm3<<<g3, 256, 0, stream>>>(x, tmk, tmv, tmr, xx, Wk, Wv, Wr, kb, vb, rb);

    const int n4m = (int)(SZM / 4);
    k_wkv<<<(n4m + 255) / 256, 256, 0, stream>>>(kb, vb, rb, tf, aa, bb, pp, zb);

    dim3 go(M_DIM / 128, C_DIM / 128);
    k_gemm_out<<<go, 256, 0, stream>>>(zb, Wo, (float*)d_out);
}
